// Block_87926570484385
// MI455X (gfx1250) — matrix operand order repaired and checked by reading, not compiled, not run on hardware
//
#include <hip/hip_runtime.h>
#include <hip/hip_bf16.h>

// ---------------------------------------------------------------------------
// Types for CDNA5 WMMA (wave32): v_wmma_f32_16x16x32_bf16
// ---------------------------------------------------------------------------
typedef __attribute__((ext_vector_type(16))) __bf16 v16bf;
typedef __attribute__((ext_vector_type(8)))  float  v8f;
typedef __attribute__((ext_vector_type(4))) unsigned u32x4;
typedef __attribute__((ext_vector_type(8))) unsigned u32x8;

struct alignas(16) Bf8 { __bf16 v[8]; };

__device__ __forceinline__ v8f wmma_bf16(v16bf a, v16bf b, v8f c) {
  return __builtin_amdgcn_wmma_f32_16x16x32_bf16(false, a, false, b, (short)0, c,
                                                 false, false);
}

__device__ __forceinline__ v8f vzero8() {
  v8f z;
#pragma unroll
  for (int i = 0; i < 8; ++i) z[i] = 0.f;
  return z;
}

// A fragment (16x32, bf16, row-major in LDS), per-lane data K-contiguous.
__device__ __forceinline__ v16bf frag_a(const __bf16* base, int lda, int lane) {
  int m = lane & 15;
  int ksel = (lane >> 4) & 1;
  const __bf16* p = base + m * lda + 16 * ksel;
  v16bf a;
#pragma unroll
  for (int i = 0; i < 16; ++i) a[i] = p[i];
  return a;
}

// B fragment from K-contiguous storage: element(k,n) = base[n*ldt + k].
__device__ __forceinline__ v16bf frag_bt(const __bf16* base, int ldt, int lane) {
  int n = lane & 15;
  int khalf = (lane >> 4) & 1;
  const __bf16* p = base + n * ldt + khalf * 16;
  v16bf b;
#pragma unroll
  for (int i = 0; i < 16; ++i) b[i] = p[i];
  return b;
}

__device__ __forceinline__ float gelu_tanh(float x) {
  float x3 = x * x * x;
  return 0.5f * x * (1.f + tanhf(0.7978845608028654f * (x + 0.044715f * x3)));
}

// LDS aperture rule: LDS byte address = low 32 bits of the flat pointer.
__device__ __forceinline__ unsigned lds_addr(const void* p) {
  return (unsigned)(size_t)p;
}

// ---------------------------------------------------------------------------
// Tensor Data Mover: DMA a 2-D bf16 tile (tileY rows x tileX elems, row stride
// strideElems) from global memory into LDS (rows packed contiguously).
// D# layout per CDNA5 ISA. 2-D: VADDR2/3 = NULL.
// Issue from one wave; completion via TENSORcnt.
// ---------------------------------------------------------------------------
__device__ __forceinline__ void tdm_load_2d(unsigned ldsAddr, const void* gptr,
                                            unsigned tileX, unsigned tileY,
                                            unsigned strideElems) {
  unsigned long long ga = (unsigned long long)(size_t)gptr;
  u32x4 g0;
  g0[0] = 1u;                                        // count=1 (valid user D#)
  g0[1] = ldsAddr;                                   // lds_addr (bytes)
  g0[2] = (unsigned)ga;                              // global_addr[31:0]
  g0[3] = ((unsigned)(ga >> 32) & 0x01FFFFFFu)       // global_addr[56:32]
          | 0x80000000u;                             // type = 2 ("image")
  u32x8 g1;
  g1[0] = 0x00010000u;                               // data_size=1 (2B); mask=0
  g1[1] = (tileX & 0xFFFFu) << 16;                   // tensor_dim0[15:0]
  g1[2] = (tileX >> 16) | ((tileY & 0xFFFFu) << 16); // dim0 hi | tensor_dim1 lo
  g1[3] = (tileY >> 16) | ((tileX & 0xFFFFu) << 16); // dim1 hi | tile_dim0
  g1[4] = tileY & 0xFFFFu;                           // tile_dim1 (tile_dim2=0)
  g1[5] = strideElems;                               // tensor_dim0_stride[31:0]
  g1[6] = 0u;
  g1[7] = 0u;
  asm volatile("tensor_load_to_lds %0, %1" :: "s"(g0), "s"(g1) : "memory");
}

__device__ __forceinline__ void tdm_wait() {
  __builtin_amdgcn_s_wait_tensorcnt((short)0);
}

// Swizzled B-weight layout (WMMA-ready 32K x 16N blocks):
//   idx(k,n) = (k/32)*(N*32) + (n/16)*512 + (n%16)*32 + (k%32)
// A K-tile's 128-column window is one contiguous 8KB slab.

// ---------------------------------------------------------------------------
// K0: time MLP  te2 = lin2(silu(lin1(te)))
// ---------------------------------------------------------------------------
__global__ void time_mlp_kernel(const float* __restrict__ te,
                                const float* __restrict__ w1,
                                const float* __restrict__ b1,
                                const float* __restrict__ w2,
                                const float* __restrict__ b2,
                                float* __restrict__ te2out) {
  __shared__ float h1[16];
  int t = threadIdx.x;
  if (t < 16) {
    float a = b1[t];
#pragma unroll 4
    for (int i = 0; i < 64; ++i) a += te[i] * w1[i * 16 + t];
    h1[t] = a / (1.f + expf(-a));  // silu
  }
  __syncthreads();
  if (t < 16) {
    float a = b2[t];
#pragma unroll
    for (int u = 0; u < 16; ++u) a += h1[u] * w2[u * 16 + t];
    te2out[t] = a;
  }
}

__global__ void ln_vec_kernel(const float* __restrict__ te2,
                              const float* __restrict__ fweight,
                              const float* __restrict__ fbias,
                              float* __restrict__ wout, float* __restrict__ bout) {
  int c = blockIdx.x * blockDim.x + threadIdx.x;
  if (c >= 768) return;
  float w = 1.f, b = 0.f;
#pragma unroll
  for (int t = 0; t < 16; ++t) {
    w += fweight[c * 16 + t] * te2[t];
    b += fbias[c * 16 + t] * te2[t];
  }
  wout[c] = w;
  bout[c] = b;
}

// ---------------------------------------------------------------------------
// Weight synthesis into the swizzled layout (coalesced reads, b128 stores)
// ---------------------------------------------------------------------------
__global__ __launch_bounds__(256) void synth_w_kernel(
    const float* __restrict__ te2, const float* __restrict__ fw,
    const float* __restrict__ fb, __bf16* __restrict__ out, int Kdim, int Ndim) {
  int tid = blockIdx.x * blockDim.x + threadIdx.x;
  int n = tid % Ndim;
  int k0 = (tid / Ndim) * 8;
  if (k0 >= Kdim) return;
  size_t sz = (size_t)Kdim * Ndim;
  float c[16];
#pragma unroll
  for (int t = 0; t < 16; ++t) c[t] = te2[t];
  Bf8 pk;
#pragma unroll
  for (int j = 0; j < 8; ++j) {
    size_t src = (size_t)(k0 + j) * Ndim + n;
    float acc = fb[src];
#pragma unroll
    for (int t = 0; t < 16; ++t) acc += c[t] * fw[(size_t)t * sz + src];
    pk.v[j] = (__bf16)acc;
  }
  size_t dst = (size_t)(k0 >> 5) * Ndim * 32 + (size_t)(n >> 4) * 512 +
               (size_t)(n & 15) * 32 + (k0 & 31);
  *(uint4*)&out[dst] = *(uint4*)&pk;
}

__global__ __launch_bounds__(256) void synth_b_kernel(
    const float* __restrict__ te2, const float* __restrict__ fbvec,
    float* __restrict__ out, int sz) {
  int i = blockIdx.x * blockDim.x + threadIdx.x;
  if (i >= sz) return;
  float acc = 0.f;
#pragma unroll
  for (int t = 0; t < 16; ++t) acc += te2[t] * fbvec[(size_t)t * sz + i];
  out[i] = acc;
}

// ---------------------------------------------------------------------------
// Temporal LayerNorm over E=768 (one block per token), bf16 output
// ---------------------------------------------------------------------------
__global__ __launch_bounds__(256) void ln_kernel(const float* __restrict__ x,
                                                 const float* __restrict__ wv,
                                                 const float* __restrict__ bv,
                                                 __bf16* __restrict__ out) {
  int row = blockIdx.x, t = threadIdx.x;
  const float* xr = x + (size_t)row * 768;
  float v0 = xr[t], v1 = xr[t + 256], v2 = xr[t + 512];
  float s = v0 + v1 + v2;
  float q = v0 * v0 + v1 * v1 + v2 * v2;
  __shared__ float rs[256], rq[256];
  rs[t] = s; rq[t] = q;
  __syncthreads();
  for (int off = 128; off > 0; off >>= 1) {
    if (t < off) { rs[t] += rs[t + off]; rq[t] += rq[t + off]; }
    __syncthreads();
  }
  float mean = rs[0] * (1.f / 768.f);
  float var  = rq[0] * (1.f / 768.f) - mean * mean;
  float inv  = rsqrtf(var + 1e-5f);
  __bf16* o = out + (size_t)row * 768;
  o[t]       = (__bf16)((v0 - mean) * inv * wv[t]       + bv[t]);
  o[t + 256] = (__bf16)((v1 - mean) * inv * wv[t + 256] + bv[t + 256]);
  o[t + 512] = (__bf16)((v2 - mean) * inv * wv[t + 512] + bv[t + 512]);
}

// ---------------------------------------------------------------------------
// bf16 GEMM with TDM double-buffered staging.
// C(MxN) = A(MxK, row-major) * B(KxN, swizzled) + bias.
// Block tile 128x128, 8 waves (4x2), each 32x64 -> 8 WMMA accumulators.
// EPI: 0 = store bf16 (+V^T side copy), 1 = gelu->bf16, 2 = f32, 3 = f32 +=
// ---------------------------------------------------------------------------
template <int EPI>
__global__ __launch_bounds__(256) void gemm_kernel(
    const __bf16* __restrict__ A, const __bf16* __restrict__ Bw,
    const float* __restrict__ bias, float* __restrict__ outF,
    __bf16* __restrict__ outB, __bf16* __restrict__ vt,
    int Mdim, int Ndim, int Kdim) {
  (void)Mdim;
  __shared__ __attribute__((aligned(16))) __bf16 sA[2][128 * 32];
  __shared__ __attribute__((aligned(16))) __bf16 sB[2][32 * 128];
  int t = threadIdx.x, lane = t & 31, w = t >> 5;
  int rowBase = blockIdx.y * 128, colBase = blockIdx.x * 128;
  int wm = w & 3, wn = w >> 2;

  v8f acc[2][4];
#pragma unroll
  for (int i = 0; i < 2; ++i)
#pragma unroll
    for (int j = 0; j < 4; ++j) acc[i][j] = vzero8();

  int nK = Kdim >> 5;
  // wave 0 drives the Tensor Data Mover
  if (w == 0) {
    tdm_load_2d(lds_addr(sA[0]), A + (size_t)rowBase * Kdim, 32, 128, (unsigned)Kdim);
    tdm_load_2d(lds_addr(sB[0]),
                Bw + (size_t)(colBase >> 4) * 512, 4096, 1, 4096);
  }
  for (int kt = 0; kt < nK; ++kt) {
    int cur = kt & 1;
    if (w == 0) tdm_wait();
    __syncthreads();  // tile kt resident; all waves done with buffer cur^1
    if (w == 0 && kt + 1 < nK) {
      const __bf16* gA = A + (size_t)rowBase * Kdim + (kt + 1) * 32;
      const __bf16* gB = Bw + (size_t)(kt + 1) * Ndim * 32 +
                         (size_t)(colBase >> 4) * 512;
      tdm_load_2d(lds_addr(sA[cur ^ 1]), gA, 32, 128, (unsigned)Kdim);
      tdm_load_2d(lds_addr(sB[cur ^ 1]), gB, 4096, 1, 4096);
    }
    v16bf a0 = frag_a(&sA[cur][(32 * wm) * 32], 32, lane);
    v16bf a1 = frag_a(&sA[cur][(32 * wm + 16) * 32], 32, lane);
#pragma unroll
    for (int nt = 0; nt < 4; ++nt) {
      int ntl = 4 * wn + nt;
      v16bf b = frag_bt(&sB[cur][ntl * 512], 32, lane);
      acc[0][nt] = wmma_bf16(a0, b, acc[0][nt]);
      acc[1][nt] = wmma_bf16(a1, b, acc[1][nt]);
    }
    __syncthreads();  // everyone done reading buffer cur before it's refilled
  }

  int r2 = lane >> 4, cl = lane & 15;
#pragma unroll
  for (int mt = 0; mt < 2; ++mt)
#pragma unroll
    for (int nt = 0; nt < 4; ++nt) {
      int tileCol = colBase + 64 * wn + 16 * nt;
      int gcol = tileCol + cl;
      Bf8 pk;
#pragma unroll
      for (int r = 0; r < 8; ++r) {
        int grow = rowBase + 32 * wm + 16 * mt + r + 8 * r2;
        float v = acc[mt][nt][r] + bias[gcol];
        size_t idx = (size_t)grow * Ndim + gcol;
        if (EPI == 0)      { __bf16 h = (__bf16)v; outB[idx] = h; pk.v[r] = h; }
        else if (EPI == 1) outB[idx] = (__bf16)gelu_tanh(v);
        else if (EPI == 2) outF[idx] = v;
        else               outF[idx] = outF[idx] + v;
      }
      if (EPI == 0 && vt != nullptr && tileCol >= 1536) {
        // V transposed side copy: vt[(b*768 + (gcol-1536)) * 2048 + s]
        int growBase = rowBase + 32 * wm + 16 * mt + 8 * r2;
        int bb = growBase >> 11, s0 = growBase & 2047;
        size_t dst = ((size_t)(bb * 768 + gcol - 1536)) * 2048 + s0;
        *(uint4*)&vt[dst] = *(uint4*)&pk;
      }
    }
}

// ---------------------------------------------------------------------------
// Causal flash attention with TDM-staged tiles.
// qkv: (B*S) x 2304 bf16 [3,H,D];  vt: V^T (B*768) x 2048;  out (B*S) x 768.
// Grid (S/64, B*H), 256 threads (8 waves).
// ---------------------------------------------------------------------------
__global__ __launch_bounds__(256) void attn_kernel(const __bf16* __restrict__ qkv,
                                                   const __bf16* __restrict__ vt,
                                                   __bf16* __restrict__ out) {
  const int S = 2048, NH = 12, LDQ = 2304;
  int qb = blockIdx.x;
  int bh = blockIdx.y;
  int b = bh / NH, h = bh % NH;
  int t = threadIdx.x, lane = t & 31, w = t >> 5;
  int wm = w & 3, wn = w >> 2;
  int r2 = lane >> 4, cl = lane & 15;

  __shared__ __attribute__((aligned(16))) __bf16 sQ[64 * 64];
  __shared__ __attribute__((aligned(16))) __bf16 sK[64 * 64];   // (key, d)
  __shared__ __attribute__((aligned(16))) __bf16 sVt[64 * 64];  // (d, key)
  __shared__ float  sS[64 * 64];
  __shared__ __attribute__((aligned(16))) __bf16 sP[64 * 64];
  __shared__ float sM[64], sL[64], sC[64];

  const __bf16* qbase = qkv + ((size_t)(b * S + qb * 64)) * LDQ + h * 64;
  if (w == 0) {
    tdm_load_2d(lds_addr(sQ), qbase, 64, 64, LDQ);
    tdm_wait();
  }
  if (t < 64) { sM[t] = -3.0e38f; sL[t] = 0.f; }

  v8f oacc[2];
  oacc[0] = vzero8(); oacc[1] = vzero8();

  for (int kb = 0; kb <= qb; ++kb) {
    __syncthreads();  // prior consumers of sK/sVt/sS/sP done (and sQ ready)
    if (w == 0) {
      const __bf16* kbase = qkv + ((size_t)(b * S + kb * 64)) * LDQ + 768 + h * 64;
      const __bf16* vtb = vt + ((size_t)(bh * 64)) * 2048 + kb * 64;
      tdm_load_2d(lds_addr(sK), kbase, 64, 64, LDQ);
      tdm_load_2d(lds_addr(sVt), vtb, 64, 64, 2048);
      tdm_wait();
    }
    __syncthreads();

    // S = Q * K^T
    v8f sacc[2];
    sacc[0] = vzero8(); sacc[1] = vzero8();
    for (int dk = 0; dk < 64; dk += 32) {
      v16bf a = frag_a(&sQ[(16 * wm) * 64 + dk], 64, lane);
#pragma unroll
      for (int nt = 0; nt < 2; ++nt) {
        v16bf bt = frag_bt(&sK[(32 * wn + 16 * nt) * 64 + dk], 64, lane);
        sacc[nt] = wmma_bf16(a, bt, sacc[nt]);
      }
    }
#pragma unroll
    for (int nt = 0; nt < 2; ++nt)
#pragma unroll
      for (int r = 0; r < 8; ++r) {
        int lrow = 16 * wm + r + 8 * r2;
        int lcol = 32 * wn + 16 * nt + cl;
        int gq = qb * 64 + lrow, gk = kb * 64 + lcol;
        float v = sacc[nt][r] * 0.125f;  // 1/sqrt(64)
        sS[lrow * 64 + lcol] = (gk <= gq) ? v : -1.0e9f;
      }
    __syncthreads();

    // online softmax (one thread per query row)
    if (t < 64) {
      float mold = sM[t];
      float mx = mold;
#pragma unroll 4
      for (int j = 0; j < 64; ++j) mx = fmaxf(mx, sS[t * 64 + j]);
      float corr = expf(mold - mx);
      float sum = 0.f;
      for (int j = 0; j < 64; ++j) {
        float p = expf(sS[t * 64 + j] - mx);
        sum += p;
        sP[t * 64 + j] = (__bf16)p;
      }
      sM[t] = mx; sC[t] = corr; sL[t] = sL[t] * corr + sum;
    }
    __syncthreads();

    // rescale running output, then O += P * V
#pragma unroll
    for (int nt = 0; nt < 2; ++nt)
#pragma unroll
      for (int r = 0; r < 8; ++r)
        oacc[nt][r] *= sC[16 * wm + r + 8 * r2];
    for (int kk = 0; kk < 64; kk += 32) {
      v16bf a = frag_a(&sP[(16 * wm) * 64 + kk], 64, lane);
#pragma unroll
      for (int nt = 0; nt < 2; ++nt) {
        v16bf bb = frag_bt(&sVt[(32 * wn + 16 * nt) * 64 + kk], 64, lane);
        oacc[nt] = wmma_bf16(a, bb, oacc[nt]);
      }
    }
  }

#pragma unroll
  for (int nt = 0; nt < 2; ++nt)
#pragma unroll
    for (int r = 0; r < 8; ++r) {
      int lrow = 16 * wm + r + 8 * r2;
      int grow = b * S + qb * 64 + lrow;
      int gcol = h * 64 + 32 * wn + 16 * nt + cl;
      out[(size_t)grow * 768 + gcol] = (__bf16)(oacc[nt][r] / sL[lrow]);
    }
}

// ---------------------------------------------------------------------------
// Host orchestration
// ---------------------------------------------------------------------------
extern "C" void kernel_launch(void* const* d_in, const int* in_sizes, int n_in,
                              void* d_out, int out_size, void* d_ws, size_t ws_size,
                              hipStream_t stream) {
  (void)in_sizes; (void)n_in; (void)out_size; (void)ws_size;
  const float* te = (const float*)d_in[0];
  const float* x  = (const float*)d_in[1];
  // d_in[2]=mask (causal, handled analytically), d_in[3]=layer_idx (unused)
  const int ALN = 4, MLN = 10, CAT = 16, APJ = 23, CFC = 30, MPJ = 37;

  size_t off = 0;
  auto take = [&](size_t bytes) -> char* {
    char* p = (char*)d_ws + off;
    off += (bytes + 255) & ~(size_t)255;
    return p;
  };
  float*  te2   = (float*)take(6 * 16 * sizeof(float));
  float*  lnw_a = (float*)take(768 * 4);
  float*  lnb_a = (float*)take(768 * 4);
  float*  lnw_m = (float*)take(768 * 4);
  float*  lnb_m = (float*)take(768 * 4);
  __bf16* Wqkv  = (__bf16*)take((size_t)768 * 2304 * 2);
  float*  bqkv  = (float*)take(2304 * 4);
  __bf16* Wap   = (__bf16*)take((size_t)768 * 768 * 2);
  float*  bap   = (float*)take(768 * 4);
  __bf16* Wfc   = (__bf16*)take((size_t)768 * 3072 * 2);
  float*  bfc   = (float*)take(3072 * 4);
  __bf16* Wmp   = (__bf16*)take((size_t)3072 * 768 * 2);
  float*  bmp   = (float*)take(768 * 4);
  __bf16* hb    = (__bf16*)take((size_t)4096 * 768 * 2);
  __bf16* h2b   = (__bf16*)take((size_t)4096 * 768 * 2);
  __bf16* qkvb  = (__bf16*)take((size_t)4096 * 2304 * 2);
  __bf16* vtb   = (__bf16*)take((size_t)2 * 768 * 2048 * 2);  // V^T
  __bf16* attnb = (__bf16*)take((size_t)4096 * 768 * 2);
  __bf16* ffb   = (__bf16*)take((size_t)4096 * 3072 * 2);

  const int groups[6] = {ALN, MLN, CAT, APJ, CFC, MPJ};
  for (int g = 0; g < 6; ++g)
    time_mlp_kernel<<<1, 32, 0, stream>>>(
        te, (const float*)d_in[groups[g] + 0], (const float*)d_in[groups[g] + 1],
        (const float*)d_in[groups[g] + 2], (const float*)d_in[groups[g] + 3],
        te2 + g * 16);

  ln_vec_kernel<<<3, 256, 0, stream>>>(te2 + 0 * 16, (const float*)d_in[ALN + 4],
                                       (const float*)d_in[ALN + 5], lnw_a, lnb_a);
  ln_vec_kernel<<<3, 256, 0, stream>>>(te2 + 1 * 16, (const float*)d_in[MLN + 4],
                                       (const float*)d_in[MLN + 5], lnw_m, lnb_m);

  auto synthW = [&](int gi, int slot, __bf16* outw, int K, int N) {
    int tasks = N * (K / 8);
    synth_w_kernel<<<(tasks + 255) / 256, 256, 0, stream>>>(
        te2 + slot * 16, (const float*)d_in[gi + 4], (const float*)d_in[gi + 5],
        outw, K, N);
  };
  synthW(CAT, 2, Wqkv, 768, 2304);
  synthW(APJ, 3, Wap, 768, 768);
  synthW(CFC, 4, Wfc, 768, 3072);
  synthW(MPJ, 5, Wmp, 3072, 768);
  auto synthB = [&](int gi, int slot, float* outb, int sz) {
    synth_b_kernel<<<(sz + 255) / 256, 256, 0, stream>>>(
        te2 + slot * 16, (const float*)d_in[gi + 6], outb, sz);
  };
  synthB(CAT, 2, bqkv, 2304);
  synthB(APJ, 3, bap, 768);
  synthB(CFC, 4, bfc, 3072);
  synthB(MPJ, 5, bmp, 768);

  ln_kernel<<<4096, 256, 0, stream>>>(x, lnw_a, lnb_a, hb);
  ln_kernel<<<4096, 256, 0, stream>>>(x, lnw_m, lnb_m, h2b);

  // qkv = h @ Wqkv + bqkv  (also emits V^T side copy)
  gemm_kernel<0><<<dim3(2304 / 128, 4096 / 128), 256, 0, stream>>>(
      hb, Wqkv, bqkv, nullptr, qkvb, vtb, 4096, 2304, 768);

  attn_kernel<<<dim3(2048 / 64, 2 * 12), 256, 0, stream>>>(qkvb, vtb, attnb);

  // attn_out = attn @ Wap + bap -> d_out (f32)
  gemm_kernel<2><<<dim3(768 / 128, 4096 / 128), 256, 0, stream>>>(
      attnb, Wap, bap, (float*)d_out, nullptr, nullptr, 4096, 768, 768);

  // ff = gelu(h2 @ Wfc + bfc)
  gemm_kernel<1><<<dim3(3072 / 128, 4096 / 128), 256, 0, stream>>>(
      h2b, Wfc, bfc, nullptr, ffb, nullptr, 4096, 3072, 768);

  // d_out += ff @ Wmp + bmp
  gemm_kernel<3><<<dim3(768 / 128, 4096 / 128), 256, 0, stream>>>(
      ffb, Wmp, bmp, (float*)d_out, nullptr, nullptr, 4096, 768, 3072);
}
